// Decoder_65103114273201
// MI455X (gfx1250) — matrix operand order repaired and checked by reading, not compiled, not run on hardware
//
#include <hip/hip_runtime.h>
#include <hip/hip_bf16.h>
#include <stddef.h>

// ---------------- types ----------------
typedef __bf16 bf16_t;
typedef __attribute__((ext_vector_type(8)))  bf16_t v8bf;
typedef __attribute__((ext_vector_type(16))) bf16_t v16bf;
typedef __attribute__((ext_vector_type(8)))  float  v8f;

#define LNUM  8
#define HNUM  16
#define DMODEL 1024
#define DKV   64
#define DFF   4096
#define VOCAB 32000
#define BSZ   4
#define SEQ   512
#define NTOK  (BSZ*SEQ)   // 2048

// ---- CDNA5 async global->LDS path (guarded; sync fallback) ----
#if defined(__has_builtin)
#  if __has_builtin(__builtin_amdgcn_global_load_async_to_lds_b128)
#    define HAVE_ASYNC_LDS 1
#  endif
#endif
#ifndef HAVE_ASYNC_LDS
#  define HAVE_ASYNC_LDS 0
#endif

__device__ __forceinline__ v16bf mk16(v8bf lo, v8bf hi) {
  v16bf r;
#pragma unroll
  for (int i = 0; i < 8; ++i) { r[i] = lo[i]; r[i + 8] = hi[i]; }
  return r;
}
__device__ __forceinline__ v8f vzero() {
  v8f z;
#pragma unroll
  for (int i = 0; i < 8; ++i) z[i] = 0.0f;
  return z;
}
__device__ __forceinline__ v8f wmma_bf16(v16bf a, v16bf b, v8f c) {
  // v_wmma_f32_16x16x32_bf16
  return __builtin_amdgcn_wmma_f32_16x16x32_bf16(false, a, false, b, (short)0, c,
                                                 false, false);
}

// copy 16 bytes global -> LDS (async DMA when available)
__device__ __forceinline__ void copy16(const bf16_t* g, bf16_t* l) {
#if HAVE_ASYNC_LDS
  typedef __attribute__((__vector_size__(16))) int v4i_t;
  __builtin_amdgcn_global_load_async_to_lds_b128(
      (__attribute__((address_space(1))) v4i_t*)g,
      (__attribute__((address_space(3))) v4i_t*)l, 0, 0);
#else
  *(v8bf*)l = *(const v8bf*)g;
#endif
}
// wait until at most N async copies outstanding
__device__ __forceinline__ void async_wait0() {
#if HAVE_ASYNC_LDS
#  if __has_builtin(__builtin_amdgcn_s_wait_asynccnt)
  __builtin_amdgcn_s_wait_asynccnt(0);
#  else
  asm volatile("s_wait_asynccnt 0x0" ::: "memory");
#  endif
#endif
}
__device__ __forceinline__ void async_wait4() {
#if HAVE_ASYNC_LDS
#  if __has_builtin(__builtin_amdgcn_s_wait_asynccnt)
  __builtin_amdgcn_s_wait_asynccnt(4);
#  else
  asm volatile("s_wait_asynccnt 0x4" ::: "memory");
#  endif
#endif
}

// ---------------- elementwise / setup kernels ----------------

__global__ void k_embed(const int* __restrict__ inp, const float* __restrict__ emb,
                        float* __restrict__ x) {
  size_t i = (size_t)blockIdx.x * 256 + threadIdx.x;   // exact grid
  int t = (int)(i >> 10);
  int d = (int)(i & 1023);
  x[i] = emb[(size_t)inp[t] * DMODEL + d];
}

__global__ void k_cvt(const float* __restrict__ in, bf16_t* __restrict__ out, size_t n) {
  for (size_t i = (size_t)blockIdx.x * 256 + threadIdx.x; i < n;
       i += (size_t)gridDim.x * 256)
    out[i] = (bf16_t)in[i];
}

// Wq/Wk/Wv (L,H,D,DK) -> bf16 (L, D, H*DK) row-major B matrices
__global__ void k_cvt_qkv(const float* __restrict__ in, bf16_t* __restrict__ out) {
  size_t i = (size_t)blockIdx.x * 256 + threadIdx.x;   // exact grid
  size_t l   = i >> 20;
  size_t rem = i & ((1u << 20) - 1);
  size_t d = rem >> 10;
  size_t c = rem & 1023;
  size_t h = c >> 6;
  size_t kk = c & 63;
  out[i] = (bf16_t)in[((l * HNUM + h) << 16) + d * 64 + kk];
}

// LayerNorm: one wave per token, bf16 output
__global__ __launch_bounds__(256) void k_ln(const float* __restrict__ x,
                                            const float* __restrict__ g,
                                            const float* __restrict__ bta,
                                            bf16_t* __restrict__ out) {
  int t = blockIdx.x * 8 + (threadIdx.x >> 5);
  int lane = threadIdx.x & 31;
  const float* xr = x + (size_t)t * DMODEL;
  float v[32];
  float s = 0.f;
#pragma unroll
  for (int i = 0; i < 32; ++i) { v[i] = xr[i * 32 + lane]; s += v[i]; }
#pragma unroll
  for (int m = 1; m < 32; m <<= 1) s += __shfl_xor(s, m, 32);
  float mu = s * (1.0f / DMODEL);
  float vs = 0.f;
#pragma unroll
  for (int i = 0; i < 32; ++i) { float d = v[i] - mu; vs += d * d; }
#pragma unroll
  for (int m = 1; m < 32; m <<= 1) vs += __shfl_xor(vs, m, 32);
  float rstd = rsqrtf(vs * (1.0f / DMODEL) + 1e-5f);
  bf16_t* orow = out + (size_t)t * DMODEL;
#pragma unroll
  for (int i = 0; i < 32; ++i) {
    int c = i * 32 + lane;
    orow[c] = (bf16_t)((v[i] - mu) * rstd * g[c] + bta[c]);
  }
}

// RoPE on fp32 Q,K -> bf16 Q,K
__global__ void k_rope(const float* __restrict__ qf, const float* __restrict__ kf,
                       bf16_t* __restrict__ qb, bf16_t* __restrict__ kb) {
  int i = blockIdx.x * 256 + threadIdx.x;  // exact grid: NTOK*HNUM*32
  int kk = i & 31;
  int h  = (i >> 5) & 15;
  int t  = i >> 9;
  int pos = t & (SEQ - 1);
  float theta = __expf(-(float)kk * (9.210340371976184f / 32.0f));
  float sn, cs;
  __sincosf((float)pos * theta, &sn, &cs);
  size_t base = (size_t)t * DMODEL + h * DKV + kk;
  float q0 = qf[base], q1 = qf[base + 32];
  qb[base]      = (bf16_t)(q0 * cs - q1 * sn);
  qb[base + 32] = (bf16_t)(q0 * sn + q1 * cs);
  float k0 = kf[base], k1 = kf[base + 32];
  kb[base]      = (bf16_t)(k0 * cs - k1 * sn);
  kb[base + 32] = (bf16_t)(k0 * sn + k1 * cs);
}

// ---------------- WMMA GEMM ----------------
// C[M,N] = A[M,K](bf16) * B[K,N](bf16) + bias.
// 256 thr / 8 waves, block tile 128x128, wave tile 32x64, K step 32.
// Double-buffered LDS fed by async global->LDS DMA (4 copies/thread/tile,
// s_wait_asynccnt 4 keeps next tile in flight during WMMA).
// MODE 0: fp32 store  1: fp32 += residual  2: bf16 store  3: bf16 relu
template <int MODE>
__global__ __launch_bounds__(256) void k_gemm(const bf16_t* __restrict__ A,
                                              const bf16_t* __restrict__ B,
                                              float* __restrict__ Cf,
                                              bf16_t* __restrict__ Cb,
                                              const float* __restrict__ bias,
                                              int K, int N) {
  __shared__ __align__(16) bf16_t As[2][128 * 32];   // [m][k]
  __shared__ __align__(16) bf16_t Bs[2][128 * 32];   // [k][n] (natural)
  const int t = threadIdx.x;
  const int wid = t >> 5, lane = t & 31, half = lane >> 4, lm = lane & 15;
  const int wm = (wid >> 1) * 32;
  const int wn = (wid & 1) * 64;
  const size_t m0 = (size_t)blockIdx.y * 128;
  const size_t n0 = (size_t)blockIdx.x * 128;

  // per-thread 16B copy coordinates: A 128x32 = 512 chunks, B 32x128 = 512 chunks
  const int ar0 = t >> 2,        ac0 = (t & 3) * 8;
  const int ar1 = ar0 + 64,      ac1 = ac0;
  const int br0 = t >> 4,        bc0 = (t & 15) * 8;
  const int br1 = br0 + 16,      bc1 = bc0;

  v8f acc[2][4];
#pragma unroll
  for (int mi = 0; mi < 2; ++mi)
#pragma unroll
    for (int ni = 0; ni < 4; ++ni) acc[mi][ni] = vzero();

  const int NK = K / 32;
  auto issue = [&](int kt, int buf) {
    const bf16_t* Ab = A + m0 * (size_t)K + (size_t)kt * 32;
    const bf16_t* Bb = B + n0 + (size_t)kt * 32 * (size_t)N;
    copy16(Ab + (size_t)ar0 * K + ac0, &As[buf][ar0 * 32 + ac0]);
    copy16(Ab + (size_t)ar1 * K + ac1, &As[buf][ar1 * 32 + ac1]);
    copy16(Bb + (size_t)br0 * N + bc0, &Bs[buf][br0 * 128 + bc0]);
    copy16(Bb + (size_t)br1 * N + bc1, &Bs[buf][br1 * 128 + bc1]);
  };

  issue(0, 0);
  int cur = 0;
  for (int kt = 0; kt < NK; ++kt) {
    if (kt + 1 < NK) { issue(kt + 1, cur ^ 1); async_wait4(); }
    else             { async_wait0(); }
    __syncthreads();   // buf[cur] ready for all waves

    v16bf afr[2], bfr[4];
#pragma unroll
    for (int mi = 0; mi < 2; ++mi) {
      const bf16_t* p = &As[cur][(wm + mi * 16 + lm) * 32];
      afr[mi] = mk16(*(const v8bf*)(p + half * 16), *(const v8bf*)(p + half * 16 + 8));
    }
#pragma unroll
    for (int ni = 0; ni < 4; ++ni) {
      const int n = wn + ni * 16 + lm;
      v16bf bv;
#pragma unroll
      for (int j = 0; j < 8; ++j) {
        bv[2 * j]     = Bs[cur][(half * 16 + 2 * j) * 128 + n];
        bv[2 * j + 1] = Bs[cur][(half * 16 + 2 * j + 1) * 128 + n];
      }
      bfr[ni] = bv;
    }
#pragma unroll
    for (int mi = 0; mi < 2; ++mi)
#pragma unroll
      for (int ni = 0; ni < 4; ++ni)
        acc[mi][ni] = wmma_bf16(afr[mi], bfr[ni], acc[mi][ni]);
    __syncthreads();   // all reads of buf[cur] done before it is refilled
    cur ^= 1;
  }

  // epilogue (C layout: lane = col lm, rows r + 8*half)
#pragma unroll
  for (int mi = 0; mi < 2; ++mi)
#pragma unroll
    for (int ni = 0; ni < 4; ++ni) {
      size_t n = n0 + wn + ni * 16 + lm;
      float bval = bias[n];
#pragma unroll
      for (int r = 0; r < 8; ++r) {
        size_t m = m0 + wm + mi * 16 + r + 8 * half;
        float v = acc[mi][ni][r] + bval;
        size_t idx = m * (size_t)N + n;
        if constexpr (MODE == 0)      Cf[idx] = v;
        else if constexpr (MODE == 1) Cf[idx] += v;
        else if constexpr (MODE == 2) Cb[idx] = (bf16_t)v;
        else                          Cb[idx] = (bf16_t)fmaxf(v, 0.0f);
      }
    }
}

// ---------------- flash attention (causal, online softmax) ----------------
__global__ __launch_bounds__(256) void k_attn(const bf16_t* __restrict__ Q,
                                              const bf16_t* __restrict__ Km,
                                              const bf16_t* __restrict__ V,
                                              bf16_t* __restrict__ Z) {
  __shared__ __align__(16) bf16_t Pst[8 * 16 * 32];
  const int t = threadIdx.x;
  const int wid = t >> 5, lane = t & 31, half = lane >> 4, lm = lane & 15;
  const int w = blockIdx.x * 8 + wid;
  const int qt = w & 31;
  const int h  = (w >> 5) & 15;
  const int b  = w >> 9;
  const int q0 = qt * 16;
  bf16_t* P = &Pst[wid * 512];

  const bf16_t* qrow = Q + ((size_t)(b * SEQ + q0 + lm) * DMODEL + h * DKV);
  v16bf aq0 = mk16(*(const v8bf*)(qrow + half * 16),
                   *(const v8bf*)(qrow + half * 16 + 8));
  v16bf aq1 = mk16(*(const v8bf*)(qrow + 32 + half * 16),
                   *(const v8bf*)(qrow + 32 + half * 16 + 8));

  v8f accz[4];
#pragma unroll
  for (int j = 0; j < 4; ++j) accz[j] = vzero();
  float row_m[8], row_s[8];
#pragma unroll
  for (int r = 0; r < 8; ++r) { row_m[r] = -1e30f; row_s[r] = 0.0f; }

  const int nkt = (q0 + 47) >> 5;
  for (int kt = 0; kt < nkt; ++kt) {
    const int kb = kt * 32;
    v8f s0 = vzero(), s1 = vzero();
    {
      const bf16_t* kr0 = Km + ((size_t)(b * SEQ + kb + lm) * DMODEL + h * DKV);
      v16bf bk = mk16(*(const v8bf*)(kr0 + half * 16),
                      *(const v8bf*)(kr0 + half * 16 + 8));
      s0 = wmma_bf16(aq0, bk, s0);
      bk = mk16(*(const v8bf*)(kr0 + 32 + half * 16),
                *(const v8bf*)(kr0 + 32 + half * 16 + 8));
      s0 = wmma_bf16(aq1, bk, s0);
      const bf16_t* kr1 = Km + ((size_t)(b * SEQ + kb + 16 + lm) * DMODEL + h * DKV);
      bk = mk16(*(const v8bf*)(kr1 + half * 16),
                *(const v8bf*)(kr1 + half * 16 + 8));
      s1 = wmma_bf16(aq0, bk, s1);
      bk = mk16(*(const v8bf*)(kr1 + 32 + half * 16),
                *(const v8bf*)(kr1 + 32 + half * 16 + 8));
      s1 = wmma_bf16(aq1, bk, s1);
    }
    float alpha[8];
#pragma unroll
    for (int r = 0; r < 8; ++r) {
      int mrow = q0 + r + 8 * half;
      float e0 = (kb + lm      <= mrow) ? s0[r] * 0.125f : -1e30f;
      float e1 = (kb + 16 + lm <= mrow) ? s1[r] * 0.125f : -1e30f;
      float tm = fmaxf(e0, e1);
#pragma unroll
      for (int m = 1; m < 16; m <<= 1) tm = fmaxf(tm, __shfl_xor(tm, m, 32));
      float nm = fmaxf(row_m[r], tm);
      float al = __expf(row_m[r] - nm);
      float p0 = __expf(e0 - nm);
      float p1 = __expf(e1 - nm);
      float rs = p0 + p1;
#pragma unroll
      for (int m = 1; m < 16; m <<= 1) rs += __shfl_xor(rs, m, 32);
      row_s[r] = row_s[r] * al + rs;
      row_m[r] = nm;
      alpha[r] = al;
      P[(r + 8 * half) * 32 + lm]      = (bf16_t)p0;
      P[(r + 8 * half) * 32 + 16 + lm] = (bf16_t)p1;
    }
#pragma unroll
    for (int j = 0; j < 4; ++j)
#pragma unroll
      for (int r = 0; r < 8; ++r) accz[j][r] *= alpha[r];

    const bf16_t* pr = &P[lm * 32];
    v16bf pa = mk16(*(const v8bf*)(pr + half * 16), *(const v8bf*)(pr + half * 16 + 8));

#pragma unroll
    for (int j = 0; j < 4; ++j) {
      const bf16_t* vp = V + ((size_t)(b * SEQ + kb) * DMODEL + h * DKV + j * 16 + lm);
      v16bf bv;
#pragma unroll
      for (int jj = 0; jj < 8; ++jj) {
        bv[2 * jj]     = vp[(size_t)(half * 16 + 2 * jj) * DMODEL];
        bv[2 * jj + 1] = vp[(size_t)(half * 16 + 2 * jj + 1) * DMODEL];
      }
      accz[j] = wmma_bf16(pa, bv, accz[j]);
    }
  }
#pragma unroll
  for (int j = 0; j < 4; ++j)
#pragma unroll
    for (int r = 0; r < 8; ++r) {
      size_t idx = (size_t)(b * SEQ + q0 + r + 8 * half) * DMODEL + h * DKV + j * 16 + lm;
      Z[idx] = (bf16_t)(accz[j][r] / row_s[r]);
    }
}

// ---------------- launch ----------------
extern "C" void kernel_launch(void* const* d_in, const int* in_sizes, int n_in,
                              void* d_out, int out_size, void* d_ws, size_t ws_size,
                              hipStream_t stream) {
  const int*   inp   = (const int*)  d_in[0];
  const float* tok   = (const float*)d_in[1];
  const float* Wq    = (const float*)d_in[2];
  const float* bq    = (const float*)d_in[3];
  const float* Wk    = (const float*)d_in[4];
  const float* bk    = (const float*)d_in[5];
  const float* Wv    = (const float*)d_in[6];
  const float* bv    = (const float*)d_in[7];
  const float* Wo    = (const float*)d_in[8];
  const float* bo    = (const float*)d_in[9];
  const float* ln1g  = (const float*)d_in[10];
  const float* ln1b  = (const float*)d_in[11];
  const float* W1    = (const float*)d_in[12];
  const float* b1    = (const float*)d_in[13];
  const float* W2    = (const float*)d_in[14];
  const float* b2    = (const float*)d_in[15];
  const float* ln2g  = (const float*)d_in[16];
  const float* ln2b  = (const float*)d_in[17];
  const float* fcw   = (const float*)d_in[18];
  const float* fcb   = (const float*)d_in[19];
  float* logits = (float*)d_out;

  char* w = (char*)d_ws;
  size_t off = 0;
  auto take = [&](size_t bytes) -> char* {
    char* p = w + off;
    off += (bytes + 255) & ~(size_t)255;
    return p;
  };
  float*  xb  = (float*) take((size_t)NTOK * DMODEL * 4);
  bf16_t* hb  = (bf16_t*)take((size_t)NTOK * DMODEL * 2);
  float*  qf  = (float*) take((size_t)NTOK * DMODEL * 4);
  float*  kf  = (float*) take((size_t)NTOK * DMODEL * 4);
  bf16_t* qbb = (bf16_t*)take((size_t)NTOK * DMODEL * 2);
  bf16_t* kbb = (bf16_t*)take((size_t)NTOK * DMODEL * 2);
  bf16_t* vbb = (bf16_t*)take((size_t)NTOK * DMODEL * 2);
  bf16_t* zbb = (bf16_t*)take((size_t)NTOK * DMODEL * 2);
  bf16_t* gbb = (bf16_t*)take((size_t)NTOK * DFF * 2);
  bf16_t* wqb = (bf16_t*)take((size_t)LNUM * DMODEL * DMODEL * 2);
  bf16_t* wkb = (bf16_t*)take((size_t)LNUM * DMODEL * DMODEL * 2);
  bf16_t* wvb = (bf16_t*)take((size_t)LNUM * DMODEL * DMODEL * 2);
  bf16_t* wob = (bf16_t*)take((size_t)LNUM * DMODEL * DMODEL * 2);
  bf16_t* w1b = (bf16_t*)take((size_t)LNUM * DMODEL * DFF * 2);
  bf16_t* w2b = (bf16_t*)take((size_t)LNUM * DFF * DMODEL * 2);
  bf16_t* fwb = (bf16_t*)take((size_t)DMODEL * VOCAB * 2);

  const size_t nQKV = (size_t)LNUM * DMODEL * DMODEL;
  k_cvt_qkv<<<(int)(nQKV / 256), 256, 0, stream>>>(Wq, wqb);
  k_cvt_qkv<<<(int)(nQKV / 256), 256, 0, stream>>>(Wk, wkb);
  k_cvt_qkv<<<(int)(nQKV / 256), 256, 0, stream>>>(Wv, wvb);
  k_cvt<<<(int)(nQKV / 256), 256, 0, stream>>>(Wo, wob, nQKV);
  const size_t nFF = (size_t)LNUM * DMODEL * DFF;
  k_cvt<<<(int)(nFF / 256), 256, 0, stream>>>(W1, w1b, nFF);
  k_cvt<<<(int)(nFF / 256), 256, 0, stream>>>(W2, w2b, nFF);
  const size_t nFC = (size_t)DMODEL * VOCAB;
  k_cvt<<<(int)(nFC / 256), 256, 0, stream>>>(fcw, fwb, nFC);

  k_embed<<<(NTOK * DMODEL) / 256, 256, 0, stream>>>(inp, tok, xb);

  dim3 blk(256);
  dim3 gSq(DMODEL / 128, NTOK / 128);
  dim3 gF1(DFF / 128, NTOK / 128);
  dim3 gF2(DMODEL / 128, NTOK / 128);
  dim3 gLg(VOCAB / 128, NTOK / 128);

  for (int l = 0; l < LNUM; ++l) {
    const size_t wOff = (size_t)l * DMODEL * DMODEL;
    const size_t fOff = (size_t)l * DMODEL * DFF;
    k_ln<<<NTOK / 8, blk, 0, stream>>>(xb, ln1g + l * DMODEL, ln1b + l * DMODEL, hb);
    k_gemm<0><<<gSq, blk, 0, stream>>>(hb, wqb + wOff, qf, nullptr, bq + l * DMODEL,
                                       DMODEL, DMODEL);
    k_gemm<0><<<gSq, blk, 0, stream>>>(hb, wkb + wOff, kf, nullptr, bk + l * DMODEL,
                                       DMODEL, DMODEL);
    k_gemm<2><<<gSq, blk, 0, stream>>>(hb, wvb + wOff, nullptr, vbb, bv + l * DMODEL,
                                       DMODEL, DMODEL);
    k_rope<<<(NTOK * HNUM * 32) / 256, blk, 0, stream>>>(qf, kf, qbb, kbb);
    k_attn<<<(BSZ * HNUM * (SEQ / 16)) / 8, blk, 0, stream>>>(qbb, kbb, vbb, zbb);
    k_gemm<1><<<gSq, blk, 0, stream>>>(zbb, wob + wOff, xb, nullptr, bo + l * DMODEL,
                                       DMODEL, DMODEL);
    k_ln<<<NTOK / 8, blk, 0, stream>>>(xb, ln2g + l * DMODEL, ln2b + l * DMODEL, hb);
    k_gemm<3><<<gF1, blk, 0, stream>>>(hb, w1b + fOff, nullptr, gbb, b1 + l * DFF,
                                       DMODEL, DFF);
    k_gemm<1><<<gF2, blk, 0, stream>>>(gbb, w2b + fOff, xb, nullptr, b2 + l * DMODEL,
                                       DFF, DMODEL);
  }

  k_cvt<<<(NTOK * DMODEL) / 256, blk, 0, stream>>>(xb, hb, (size_t)NTOK * DMODEL);
  k_gemm<0><<<gLg, blk, 0, stream>>>(hb, fwb, logits, nullptr, fcb, DMODEL, VOCAB);
}
